// ResUNet_21758304321758
// MI455X (gfx1250) — matrix operand order repaired and checked by reading, not compiled, not run on hardware
//
#include <hip/hip_runtime.h>
#include <stdint.h>
#include <stddef.h>

#define NVOX 120000
#define KTAPS 27

typedef __attribute__((ext_vector_type(16))) __bf16 v16bf;
typedef __attribute__((ext_vector_type(8)))  __bf16 v8bf;
typedef __attribute__((ext_vector_type(8)))  float  v8f;

// ---------- bf16 helpers (bit-level, buffers are uint16_t) ----------
static __device__ __forceinline__ uint16_t f2bf_bits(float f) {
  uint32_t u = __builtin_bit_cast(uint32_t, f);
  uint32_t r = u + 0x7FFFu + ((u >> 16) & 1u);   // round-to-nearest-even
  return (uint16_t)(r >> 16);
}
static __device__ __forceinline__ float bf2f_bits(uint16_t h) {
  uint32_t u = ((uint32_t)h) << 16;
  return __builtin_bit_cast(float, u);
}

// =====================================================================================
// Fused sparse-conv GEMM:  Y[i,:] = epilogue( sum_k  X[nbr[i,k],:] @ W[k] )
// One wave computes a 16 x (CT*16) output tile: the gathered A fragment (per-lane 16B
// loads matching the CDNA5 16-bit A layout) is reused across CT WMMAs, and the 4 waves
// of a block cover 4 adjacent row tiles with the SAME column group so the repacked B
// blob stays L0-resident. Epilogue: y = acc*s + t (+residual)(+relu) -> bf16/fp32.
// =====================================================================================
template <int CT>
__global__ __launch_bounds__(128) void sconv_wmma_kernel(
    const uint16_t* __restrict__ X,       // [n, cinPad] bf16 bits
    const int*      __restrict__ nbr,     // [n, 27] or nullptr => identity gather
    const uint32_t* __restrict__ Wp,      // packed: (((k*cCh+c)*colTiles+ct)*32+lane)*8 dwords
    const float*    __restrict__ scale,   // [cout] or nullptr (=> 1)
    const float*    __restrict__ shift,   // [cout] or nullptr (=> 0)
    const uint16_t* __restrict__ res,     // residual [n, cout] bf16 or nullptr
    uint16_t*       __restrict__ Yb,      // bf16 out or nullptr
    float*          __restrict__ Yf,      // fp32 out or nullptr
    int n, int cinPad, int cout, int kTaps, int doRelu)
{
  const int lane     = threadIdx.x & 31;
  const int wave     = threadIdx.x >> 5;
  const int colTiles = cout >> 4;
  const int ct0      = blockIdx.y * CT;           // first of CT column tiles
  const int rowTile  = blockIdx.x * 4 + wave;     // 4 row tiles per block
  if (rowTile * 16 >= n) return;
  const int m       = lane & 15;
  const int hi      = lane >> 4;
  const int row     = rowTile * 16 + m;           // A-fragment row for this lane
  const int cChunks = cinPad >> 5;

  v8f acc[CT];
#pragma unroll
  for (int i = 0; i < CT; ++i) acc[i] = (v8f){};

  const uint32_t* wlane = Wp + lane * 8;
  for (int k = 0; k < kTaps; ++k) {
    const int idx    = nbr ? nbr[row * KTAPS + k] : row;
    const bool valid = (idx < n);
    const uint16_t* xrow = X + (size_t)idx * cinPad + hi * 8;
    for (int c = 0; c < cChunks; ++c) {
      v16bf a = {};
      if (valid) {
        v8bf lo  = *(const v8bf*)(const void*)(xrow + c * 32);       // K = hi*8 .. +7
        v8bf hiv = *(const v8bf*)(const void*)(xrow + c * 32 + 16);  // K = 16+hi*8 .. +7
        a = __builtin_shufflevector(lo, hiv, 0,1,2,3,4,5,6,7,8,9,10,11,12,13,14,15);
      }
      const uint32_t* bp = wlane + (size_t)((k * cChunks + c) * colTiles + ct0) * 256;
#pragma unroll
      for (int i = 0; i < CT; ++i) {
        v16bf b = *(const v16bf*)(const void*)(bp + i * 256);
        acc[i] = __builtin_amdgcn_wmma_f32_16x16x32_bf16(
            /*neg_a=*/false, a, /*neg_b=*/false, b,
            /*c_mod=*/(short)0, acc[i], /*reuse_a=*/false, /*reuse_b=*/false);
      }
    }
  }

  // C/D layout: lane -> col = ct*16 + (lane&15); vgpr v -> row = 16*tile + v + 8*hi
#pragma unroll
  for (int i = 0; i < CT; ++i) {
    const int col = (ct0 + i) * 16 + m;
    const float s = scale ? scale[col] : 1.0f;
    const float t = shift ? shift[col] : 0.0f;
#pragma unroll
    for (int v = 0; v < 8; ++v) {
      const int r = rowTile * 16 + hi * 8 + v;
      if (r < n) {
        float y = acc[i][v] * s + t;
        if (res)    y += bf2f_bits(res[(size_t)r * cout + col]);
        if (doRelu) y = fmaxf(y, 0.0f);
        const size_t o = (size_t)r * cout + col;
        if (Yb) Yb[o] = f2bf_bits(y);
        if (Yf) Yf[o] = y;
      }
    }
  }
}

// ---------- weight repack: fp32 [k][cinReal][cout] -> bf16 fragment-native blob ------
__global__ void repack_weights(const float* __restrict__ W, uint32_t* __restrict__ Wp,
                               int kTaps, int cinReal, int cinPad, int cout)
{
  const int total = kTaps * (cinPad >> 5) * (cout >> 4) * 256;
  const int tid = blockIdx.x * blockDim.x + threadIdx.x;
  if (tid >= total) return;
  const int j    = tid & 7;
  const int lane = (tid >> 3) & 31;
  int rest       = tid >> 8;
  const int colTiles = cout >> 4;
  const int ct = rest % colTiles; rest /= colTiles;
  const int cCh = cinPad >> 5;
  const int c  = rest % cCh;
  const int k  = rest / cCh;
  const int nl = lane & 15, hi = lane >> 4;
  const int K0 = c * 32 + hi * 8 + (j >> 2) * 16 + 2 * (j & 3);      // B layout: lanes0-15 K=0..7,16..23; 16-31 K=8..15,24..31
  const int col = ct * 16 + nl;
  const float w0 = (K0     < cinReal) ? W[((size_t)k * cinReal + K0    ) * cout + col] : 0.0f;
  const float w1 = (K0 + 1 < cinReal) ? W[((size_t)k * cinReal + K0 + 1) * cout + col] : 0.0f;
  Wp[tid] = (uint32_t)f2bf_bits(w0) | ((uint32_t)f2bf_bits(w1) << 16);
}

// ---------- BN fold: s = g*rsqrt(v+eps); t = b - m*s ----------
__global__ void bn_prep(const float* __restrict__ g, const float* __restrict__ b,
                        const float* __restrict__ m, const float* __restrict__ v,
                        float* __restrict__ s, float* __restrict__ t, int c)
{
  const int i = blockIdx.x * blockDim.x + threadIdx.x;
  if (i < c) {
    const float sc = g[i] * rsqrtf(v[i] + 1e-5f);
    s[i] = sc;
    t[i] = b[i] - m[i] * sc;
  }
}

// ---------- input: fp32 [N,3] -> bf16 [N,32] zero-padded ----------
__global__ void pad_convert_x(const float* __restrict__ x, uint16_t* __restrict__ X0, int n)
{
  const int tid = blockIdx.x * blockDim.x + threadIdx.x;
  if (tid < n * 32) {
    const int r = tid >> 5, c = tid & 31;
    X0[tid] = f2bf_bits(c < 3 ? x[r * 3 + c] : 0.0f);
  }
}

// ---------- channel concat (bf16) ----------
__global__ void concat_bf16(const uint16_t* __restrict__ a, int c1,
                            const uint16_t* __restrict__ b, int c2,
                            uint16_t* __restrict__ dst, int n)
{
  const int C = c1 + c2;
  const int tid = blockIdx.x * blockDim.x + threadIdx.x;
  if (tid < n * C) {
    const int r = tid / C, c = tid - r * C;
    dst[tid] = (c < c1) ? a[(size_t)r * c1 + c] : b[(size_t)r * c2 + (c - c1)];
  }
}

// =====================================================================================
// Host orchestration
// =====================================================================================
namespace {

struct ConvSpec { int src, kT, cinR, cinP, cout; };
struct BnSpec   { int b, g, m, v, C; };

// JAX pytree flatten (dicts alphabetical at every level):
// 0:nbr | block1(1..10) block2(11..20) block2_tr(21..30) block3(31..40)
// block3_tr(41..50) block4(51..60) block4_tr(61..70) | bn1(71..74) bn2(75..78)
// bn2_tr(79..82) bn3(83..86) bn3_tr(87..90) bn4(91..94) bn4_tr(95..98) |
// conv1(99) conv1_tr(100) conv2(101) conv2_tr(102) conv3(103) conv3_tr(104)
// conv4(105) conv4_tr(106) final_b(107) final_w(108) | x(109)
// Each block: bn1{b,g,m,v} bn2{b,g,m,v} w1 w2.
static const ConvSpec kConvs[23] = {
  { 99, 27,   3,  32,  32},  // 0 conv1
  {  9, 27,  32,  32,  32},  // 1 block1.w1
  { 10, 27,  32,  32,  32},  // 2 block1.w2
  {101, 27,  32,  32,  64},  // 3 conv2
  { 19, 27,  64,  64,  64},  // 4 block2.w1
  { 20, 27,  64,  64,  64},  // 5 block2.w2
  {103, 27,  64,  64, 128},  // 6 conv3
  { 39, 27, 128, 128, 128},  // 7 block3.w1
  { 40, 27, 128, 128, 128},  // 8 block3.w2
  {105, 27, 128, 128, 256},  // 9 conv4
  { 59, 27, 256, 256, 256},  // 10 block4.w1
  { 60, 27, 256, 256, 256},  // 11 block4.w2
  {106, 27, 256, 256, 128},  // 12 conv4_tr
  { 69, 27, 128, 128, 128},  // 13 block4_tr.w1
  { 70, 27, 128, 128, 128},  // 14 block4_tr.w2
  {104, 27, 256, 256,  64},  // 15 conv3_tr
  { 49, 27,  64,  64,  64},  // 16 block3_tr.w1
  { 50, 27,  64,  64,  64},  // 17 block3_tr.w2
  {102, 27, 128, 128,  64},  // 18 conv2_tr
  { 29, 27,  64,  64,  64},  // 19 block2_tr.w1
  { 30, 27,  64,  64,  64},  // 20 block2_tr.w2
  {100, 27,  96,  96,  64},  // 21 conv1_tr
  {108,  1,  64,  64,  32},  // 22 final (identity gather)
};

static const BnSpec kBns[21] = {
  {71,72,73,74, 32},  // 0  bn1
  { 1, 2, 3, 4, 32},  // 1  block1.bn1
  { 5, 6, 7, 8, 32},  // 2  block1.bn2
  {75,76,77,78, 64},  // 3  bn2
  {11,12,13,14, 64},  // 4  block2.bn1
  {15,16,17,18, 64},  // 5  block2.bn2
  {83,84,85,86,128},  // 6  bn3
  {31,32,33,34,128},  // 7  block3.bn1
  {35,36,37,38,128},  // 8  block3.bn2
  {91,92,93,94,256},  // 9  bn4
  {51,52,53,54,256},  // 10 block4.bn1
  {55,56,57,58,256},  // 11 block4.bn2
  {95,96,97,98,128},  // 12 bn4_tr
  {61,62,63,64,128},  // 13 block4_tr.bn1
  {65,66,67,68,128},  // 14 block4_tr.bn2
  {87,88,89,90, 64},  // 15 bn3_tr
  {41,42,43,44, 64},  // 16 block3_tr.bn1
  {45,46,47,48, 64},  // 17 block3_tr.bn2
  {79,80,81,82, 64},  // 18 bn2_tr
  {21,22,23,24, 64},  // 19 block2_tr.bn1
  {25,26,27,28, 64},  // 20 block2_tr.bn2
};

static inline size_t align256(size_t x) { return (x + 255) & ~(size_t)255; }

} // namespace

extern "C" void kernel_launch(void* const* d_in, const int* in_sizes, int n_in,
                              void* d_out, int out_size, void* d_ws, size_t ws_size,
                              hipStream_t stream) {
  (void)in_sizes; (void)n_in; (void)out_size; (void)ws_size;
  const int n = NVOX;
  const int* nbr = (const int*)d_in[0];
  const float* x = (const float*)d_in[109];

  // ---- workspace bump allocator ----
  char* ws = (char*)d_ws;
  size_t off = 0;
  auto alloc = [&](size_t bytes) -> void* {
    void* p = ws + off;
    off = align256(off + bytes);
    return p;
  };

  uint16_t* W0 = (uint16_t*)alloc((size_t)n * 256 * 2);
  uint16_t* W1 = (uint16_t*)alloc((size_t)n * 256 * 2);
  uint16_t* W2 = (uint16_t*)alloc((size_t)n * 256 * 2);
  uint16_t* S1 = (uint16_t*)alloc((size_t)n *  32 * 2);
  uint16_t* S2 = (uint16_t*)alloc((size_t)n *  64 * 2);
  uint16_t* S4 = (uint16_t*)alloc((size_t)n * 128 * 2);

  uint32_t* wp[23];
  for (int i = 0; i < 23; ++i) {
    const ConvSpec& cs = kConvs[i];
    const size_t dwords = (size_t)cs.kT * (cs.cinP >> 5) * (cs.cout >> 4) * 256;
    wp[i] = (uint32_t*)alloc(dwords * 4);
  }
  float* bs[21]; float* bt[21];
  for (int i = 0; i < 21; ++i) {
    bs[i] = (float*)alloc(kBns[i].C * 4);
    bt[i] = (float*)alloc(kBns[i].C * 4);
  }

  // ---- per-call preprocessing (weights repack, BN fold, input pad) ----
  for (int i = 0; i < 23; ++i) {
    const ConvSpec& cs = kConvs[i];
    const int total = cs.kT * (cs.cinP >> 5) * (cs.cout >> 4) * 256;
    repack_weights<<<(total + 255) / 256, 256, 0, stream>>>(
        (const float*)d_in[cs.src], wp[i], cs.kT, cs.cinR, cs.cinP, cs.cout);
  }
  for (int i = 0; i < 21; ++i) {
    const BnSpec& b = kBns[i];
    bn_prep<<<(b.C + 63) / 64, 64, 0, stream>>>(
        (const float*)d_in[b.g], (const float*)d_in[b.b],
        (const float*)d_in[b.m], (const float*)d_in[b.v], bs[i], bt[i], b.C);
  }
  pad_convert_x<<<(n * 32 + 255) / 256, 256, 0, stream>>>(x, W0, n);

  // ---- fused conv launcher (wave = 16x64 tile when cout>=64, else 16x32) ----
  auto sconv = [&](const uint16_t* X, const int* nb, int ci, // conv idx
                   const float* s, const float* t,
                   const uint16_t* res, uint16_t* Yb, float* Yf, int relu) {
    const ConvSpec& cs = kConvs[ci];
    const int colTiles = cs.cout >> 4;
    const int rowBlks  = (n / 16 + 3) / 4;        // 4 row tiles (waves) per block
    if (colTiles >= 4) {
      dim3 grid(rowBlks, colTiles / 4);
      sconv_wmma_kernel<4><<<grid, 128, 0, stream>>>(
          X, nb, wp[ci], s, t, res, Yb, Yf, n, cs.cinP, cs.cout, cs.kT, relu);
    } else {
      dim3 grid(rowBlks, 1);
      sconv_wmma_kernel<2><<<grid, 128, 0, stream>>>(
          X, nb, wp[ci], s, t, res, Yb, Yf, n, cs.cinP, cs.cout, cs.kT, relu);
    }
  };
  auto cat = [&](const uint16_t* a, int c1, const uint16_t* b, int c2, uint16_t* dst) {
    const int total = n * (c1 + c2);
    concat_bf16<<<(total + 255) / 256, 256, 0, stream>>>(a, c1, b, c2, dst, n);
  };

  // ---- forward pipeline (P = bn(conv(X)); A = relu(bn(conv(P))); O = relu(P + bn(conv(A)))) ----
  // stage1 (32)
  sconv(W0, nbr, 0, bs[0], bt[0], nullptr, W1, nullptr, 0);
  sconv(W1, nbr, 1, bs[1], bt[1], nullptr, W2, nullptr, 1);
  sconv(W2, nbr, 2, bs[2], bt[2], W1, S1, nullptr, 1);
  // stage2 (64)
  sconv(S1, nbr, 3, bs[3], bt[3], nullptr, W1, nullptr, 0);
  sconv(W1, nbr, 4, bs[4], bt[4], nullptr, W2, nullptr, 1);
  sconv(W2, nbr, 5, bs[5], bt[5], W1, S2, nullptr, 1);
  // stage3 (128)
  sconv(S2, nbr, 6, bs[6], bt[6], nullptr, W1, nullptr, 0);
  sconv(W1, nbr, 7, bs[7], bt[7], nullptr, W2, nullptr, 1);
  sconv(W2, nbr, 8, bs[8], bt[8], W1, S4, nullptr, 1);
  // stage4 (256) -> s8 in W0
  sconv(S4, nbr, 9,  bs[9],  bt[9],  nullptr, W1, nullptr, 0);
  sconv(W1, nbr, 10, bs[10], bt[10], nullptr, W2, nullptr, 1);
  sconv(W2, nbr, 11, bs[11], bt[11], W1, W0, nullptr, 1);
  // stage4_tr (256 -> 128) -> o4t in W0
  sconv(W0, nbr, 12, bs[12], bt[12], nullptr, W1, nullptr, 0);
  sconv(W1, nbr, 13, bs[13], bt[13], nullptr, W2, nullptr, 1);
  sconv(W2, nbr, 14, bs[14], bt[14], W1, W0, nullptr, 1);
  cat(W0, 128, S4, 128, W2);                       // 256-ch concat in W2
  // stage3_tr (256 -> 64) -> o3t in W2
  sconv(W2, nbr, 15, bs[15], bt[15], nullptr, W1, nullptr, 0);
  sconv(W1, nbr, 16, bs[16], bt[16], nullptr, W0, nullptr, 1);
  sconv(W0, nbr, 17, bs[17], bt[17], W1, W2, nullptr, 1);
  cat(W2, 64, S2, 64, W0);                         // 128-ch concat in W0
  // stage2_tr (128 -> 64) -> o2t in W0
  sconv(W0, nbr, 18, bs[18], bt[18], nullptr, W1, nullptr, 0);
  sconv(W1, nbr, 19, bs[19], bt[19], nullptr, W2, nullptr, 1);
  sconv(W2, nbr, 20, bs[20], bt[20], W1, W0, nullptr, 1);
  cat(W0, 64, S1, 32, W2);                         // 96-ch concat in W2
  // conv1_tr (96 -> 64), relu only
  sconv(W2, nbr, 21, nullptr, nullptr, nullptr, W0, nullptr, 1);
  // final 1x1 (64 -> 32), identity gather, bias as shift, fp32 straight to d_out
  sconv(W0, nullptr, 22, nullptr, (const float*)d_in[107], nullptr, nullptr,
        (float*)d_out, 0);
}
